// A_9835475108252
// MI455X (gfx1250) — matrix operand order repaired and checked by reading, not compiled, not run on hardware
//
#include <hip/hip_runtime.h>
#include <math.h>

// ---------------------------------------------------------------------------
// MI455X (gfx1250) implementation: everything heavy goes through one tiled
// bf16 WMMA GEMM (v_wmma_f32_16x16x32_bf16), f32 accumulate. wave32, WGP.
// Block tile 128x64, 8 waves, each wave owns a 32x32 macro-tile (4 WMMAs /
// K-step). Interior tiles stream through vectorized b64 global loads +
// global_prefetch; edges fall back to guarded scalar loads.
// ---------------------------------------------------------------------------

typedef unsigned short ushort_t;
typedef __attribute__((ext_vector_type(16))) __bf16       v16bf;
typedef __attribute__((ext_vector_type(8)))  float        v8f;
typedef __attribute__((ext_vector_type(8)))  unsigned int v8u;
typedef __attribute__((ext_vector_type(2)))  unsigned int u32x2;

static constexpr int B_   = 64;
static constexpr int H_   = 768;
static constexpr int LA_  = 272;   // a-branch sequence length
static constexpr int ML_  = 256;   // b-branch sequence length
static constexpr int OUTC = 2003;  // 3 + 1000 + 1000 output columns

// ---- bf16 helpers (storage = ushort, math = f32) --------------------------
__device__ __forceinline__ ushort_t f2bf(float f) {
  unsigned int u = __float_as_uint(f);
  u += 0x7FFFu + ((u >> 16) & 1u);           // round-to-nearest-even
  return (ushort_t)(u >> 16);
}
__device__ __forceinline__ float bf2f(ushort_t h) {
  return __uint_as_float(((unsigned int)h) << 16);
}

// ---------------------------------------------------------------------------
// Generic batched GEMM:  C[b] = alpha * op(A[b]) @ op(B[b]) + bias
//   TA: op(A)[m][k] = A[k*lda + m]   (else A[m*lda + k])
//   TB: op(B)[k][n] = B[n*ldb + k]   (else B[k*ldb + n])
//   OUTBF: write bf16, else f32. bias (f32, per-column n) may be null.
// ---------------------------------------------------------------------------
static constexpr int BM   = 128;
static constexpr int BN   = 64;
static constexpr int LDAP = 36;  // LDS pitch for A tile (128 x 32), 8B aligned rows
static constexpr int LDBP = 68;  // LDS pitch for B tile (32 x 64),  8B aligned rows

template <bool TA, bool TB, bool OUTBF>
__global__ __launch_bounds__(256) void k_gemm(
    const ushort_t* __restrict__ A, long lda, long sA,
    const ushort_t* __restrict__ Bm, long ldb, long sB,
    const float* __restrict__ bias,
    void* __restrict__ Cv, long ldc, long sC,
    int M, int N, int K, float alpha) {
  __shared__ ushort_t lsA[2][BM * LDAP];
  __shared__ ushort_t lsB[2][32 * LDBP];

  const int tid  = threadIdx.x;
  const int lane = tid & 31;
  const int wid  = tid >> 5;
  const int wm   = wid & 3;   // 4 row-slabs of 32
  const int wn   = wid >> 2;  // 2 col-slabs of 32
  const int lm   = lane & 15;
  const int hi   = lane >> 4;

  const int bm0 = blockIdx.x * BM;
  const int bn0 = blockIdx.y * BN;

  A  += (long)blockIdx.z * sA;
  Bm += (long)blockIdx.z * sB;

  const bool alA = ((lda & 3) == 0) && ((((size_t)A) & 7) == 0);
  const bool alB = ((ldb & 3) == 0) && ((((size_t)Bm) & 7) == 0);

  const int nk = (K + 31) >> 5;

  auto loadTile = [&](int kt, int buf) {
    const int k0 = kt << 5;
    const bool fk = (k0 + 32 <= K);
    // ---------------- A tile: 128 (m) x 32 (k) ----------------
    if (!TA) {
      if (fk && alA && (bm0 + BM <= M)) {
        #pragma unroll
        for (int j = 0; j < 4; ++j) {
          int q = tid + j * 256;
          int m = q >> 3, kk = (q & 7) * 4;
          const ushort_t* g = A + (long)(bm0 + m) * lda + (k0 + kk);
          if (j == 0) __builtin_prefetch(g + 64, 0, 1);  // K-tile after next
          *(u32x2*)&lsA[buf][m * LDAP + kk] = *(const u32x2*)g;
        }
      } else {
        for (int i = tid; i < BM * 32; i += 256) {
          int m = i >> 5, kk = i & 31;
          int gm = bm0 + m, gk = k0 + kk;
          ushort_t v = 0;
          if (gm < M && gk < K) v = A[(long)gm * lda + gk];
          lsA[buf][m * LDAP + kk] = v;
        }
      }
    } else {
      if (fk && alA && (bm0 + BM <= M)) {
        #pragma unroll
        for (int j = 0; j < 4; ++j) {
          int q = tid + j * 256;
          int kk = q >> 5, m = (q & 31) * 4;
          const ushort_t* g = A + (long)(k0 + kk) * lda + (bm0 + m);
          union { u32x2 w; ushort_t e[4]; } u;
          u.w = *(const u32x2*)g;
          ushort_t* s = &lsA[buf][m * LDAP + kk];
          s[0] = u.e[0]; s[LDAP] = u.e[1]; s[2 * LDAP] = u.e[2]; s[3 * LDAP] = u.e[3];
        }
      } else {
        for (int i = tid; i < BM * 32; i += 256) {
          int kk = i >> 7, m = i & 127;
          int gm = bm0 + m, gk = k0 + kk;
          ushort_t v = 0;
          if (gm < M && gk < K) v = A[(long)gk * lda + gm];
          lsA[buf][m * LDAP + kk] = v;
        }
      }
    }
    // ---------------- B tile: 32 (k) x 64 (n) ----------------
    if (!TB) {
      if (fk && alB && (bn0 + BN <= N)) {
        #pragma unroll
        for (int j = 0; j < 2; ++j) {
          int q = tid + j * 256;
          int k = q >> 4, n = (q & 15) * 4;
          const ushort_t* g = Bm + (long)(k0 + k) * ldb + (bn0 + n);
          if (j == 0) __builtin_prefetch(g + 64 * ldb, 0, 1);
          *(u32x2*)&lsB[buf][k * LDBP + n] = *(const u32x2*)g;
        }
      } else {
        for (int i = tid; i < 32 * 64; i += 256) {
          int k = i >> 6, n = i & 63;
          int gk = k0 + k, gn = bn0 + n;
          ushort_t v = 0;
          if (gk < K && gn < N) v = Bm[(long)gk * ldb + gn];
          lsB[buf][k * LDBP + n] = v;
        }
      }
    } else {
      if (fk && alB && (bn0 + BN <= N)) {
        #pragma unroll
        for (int j = 0; j < 2; ++j) {
          int q = tid + j * 256;
          int n = q >> 3, k = (q & 7) * 4;
          const ushort_t* g = Bm + (long)(bn0 + n) * ldb + (k0 + k);
          union { u32x2 w; ushort_t e[4]; } u;
          u.w = *(const u32x2*)g;
          ushort_t* s = &lsB[buf][k * LDBP + n];
          s[0] = u.e[0]; s[LDBP] = u.e[1]; s[2 * LDBP] = u.e[2]; s[3 * LDBP] = u.e[3];
        }
      } else {
        for (int i = tid; i < 32 * 64; i += 256) {
          int n = i >> 5, k = i & 31;
          int gk = k0 + k, gn = bn0 + n;
          ushort_t v = 0;
          if (gk < K && gn < N) v = Bm[(long)gn * ldb + gk];
          lsB[buf][k * LDBP + n] = v;
        }
      }
    }
  };

  v8f acc[2][2] = {};

  int buf = 0;
  loadTile(0, 0);
  __syncthreads();

  for (int kt = 0; kt < nk; ++kt) {
    if (kt + 1 < nk) loadTile(kt + 1, buf ^ 1);

    // Fragments per ISA 7.12.2: A lane<16 rows m, K 0-7 & 16-23; lane>=16
    // K 8-15 & 24-31. B lane = n column, same K order as A. 2 A-frags x 2 B-frags.
    v16bf afr[2], bfr[2];
    #pragma unroll
    for (int s = 0; s < 2; ++s) {
      const ushort_t* As = &lsA[buf][(wm * 32 + s * 16 + lm) * LDAP];
      v8u au;
      #pragma unroll
      for (int p = 0; p < 8; ++p) {
        int kk = ((p < 4) ? (p * 2) : (16 + (p - 4) * 2)) + hi * 8;
        au[p] = *(const unsigned int*)(As + kk);
      }
      afr[s] = __builtin_bit_cast(v16bf, au);
      const ushort_t* Bs = &lsB[buf][wn * 32 + s * 16 + lm];
      v8u bu;
      #pragma unroll
      for (int p = 0; p < 8; ++p) {
        int kk = ((p < 4) ? (p * 2) : (16 + (p - 4) * 2)) + hi * 8;
        bu[p] = (unsigned int)Bs[kk * LDBP] | ((unsigned int)Bs[(kk + 1) * LDBP] << 16);
      }
      bfr[s] = __builtin_bit_cast(v16bf, bu);
    }
    #pragma unroll
    for (int si = 0; si < 2; ++si)
      #pragma unroll
      for (int sj = 0; sj < 2; ++sj)
        acc[si][sj] = __builtin_amdgcn_wmma_f32_16x16x32_bf16(
            false, afr[si], false, bfr[sj], (short)0, acc[si][sj], false, false);

    __syncthreads();
    buf ^= 1;
  }

  // ---- store D: row = bm0+wm*32+si*16+r+8*hi, col = bn0+wn*32+sj*16+lm ----
  Cv = (void*)((char*)Cv +
               (long)blockIdx.z * sC * (OUTBF ? sizeof(ushort_t) : sizeof(float)));
  #pragma unroll
  for (int si = 0; si < 2; ++si) {
    #pragma unroll
    for (int sj = 0; sj < 2; ++sj) {
      int col = bn0 + wn * 32 + sj * 16 + lm;
      if (col >= N) continue;
      float bv = bias ? bias[col] : 0.f;
      #pragma unroll
      for (int r = 0; r < 8; ++r) {
        int row = bm0 + wm * 32 + si * 16 + r + 8 * hi;
        if (row >= M) continue;
        float v = alpha * acc[si][sj][r] + bv;
        if (OUTBF)
          ((ushort_t*)Cv)[(long)row * ldc + col] = f2bf(v);
        else
          ((float*)Cv)[(long)row * ldc + col] = v;
      }
    }
  }
}

// ---------------------------------------------------------------------------
// Elementwise / prep kernels
// ---------------------------------------------------------------------------
__global__ void k_cast(const float* __restrict__ s, ushort_t* __restrict__ d, long n) {
  long i = (long)blockIdx.x * blockDim.x + threadIdx.x;
  if (i < n) d[i] = f2bf(s[i]);
}

// src: (B, L+1, 768) f32 -> x: (B, L, 768) bf16 and xt: (B, 768, L) bf16,
// skipping the CLS row.
__global__ void k_cast_xt(const float* __restrict__ src, ushort_t* __restrict__ x,
                          ushort_t* __restrict__ xt, int L, long n) {
  long i = (long)blockIdx.x * blockDim.x + threadIdx.x;
  if (i >= n) return;
  int d = (int)(i % H_);
  long t = i / H_;
  int l = (int)(t % L);
  int b = (int)(t / L);
  float v = src[((long)b * (L + 1) + (l + 1)) * H_ + d];
  ushort_t h = f2bf(v);
  x[i] = h;
  xt[((long)b * H_ + d) * L + l] = h;
}

// h[b] = concat(tokens[b,0,:], cls_tokens[b,0,:])  -> (B, 1536) bf16
__global__ void k_build_h(const float* __restrict__ tok, const float* __restrict__ cls,
                          ushort_t* __restrict__ h) {
  int i = blockIdx.x * blockDim.x + threadIdx.x;
  if (i >= B_ * 2 * H_) return;
  int b = i / (2 * H_), c = i % (2 * H_);
  float v = (c < H_) ? tok[(long)b * (LA_ + 1) * H_ + c]
                     : cls[(long)b * (ML_ + 1) * H_ + (c - H_)];
  h[i] = f2bf(v);
}

// In-place row softmax on bf16 data, one 256-thread block per row (len<=768).
__global__ void k_softmax(ushort_t* __restrict__ data, int len) {
  __shared__ float red[256];
  long base = (long)blockIdx.x * len;
  int t = threadIdx.x;
  float mx = -3.0e38f;
  for (int i = t; i < len; i += 256) mx = fmaxf(mx, bf2f(data[base + i]));
  red[t] = mx; __syncthreads();
  for (int s = 128; s > 0; s >>= 1) {
    if (t < s) red[t] = fmaxf(red[t], red[t + s]);
    __syncthreads();
  }
  mx = red[0]; __syncthreads();
  float vals[4];
  float sum = 0.f;
  int cnt = 0;
  for (int i = t; i < len; i += 256) {
    float e = __expf(bf2f(data[base + i]) - mx);
    vals[cnt++] = e;
    sum += e;
  }
  red[t] = sum; __syncthreads();
  for (int s = 128; s > 0; s >>= 1) {
    if (t < s) red[t] += red[t + s];
    __syncthreads();
  }
  float inv = 1.0f / red[0];
  cnt = 0;
  for (int i = t; i < len; i += 256) data[base + i] = f2bf(vals[cnt++] * inv);
}

// ---------------------------------------------------------------------------
// Host-side dispatch
// ---------------------------------------------------------------------------
static inline void gemm(hipStream_t st, bool TAf, bool TBf, bool OBF,
                        const ushort_t* A, long lda, long sA,
                        const ushort_t* Bm, long ldb, long sB,
                        const float* bias, void* C, long ldc, long sC,
                        int M, int N, int K, int nb, float alpha) {
  dim3 g((M + BM - 1) / BM, (N + BN - 1) / BN, nb), blk(256);
  if (!TAf && !TBf && OBF)
    k_gemm<false, false, true><<<g, blk, 0, st>>>(A, lda, sA, Bm, ldb, sB, bias, C, ldc, sC, M, N, K, alpha);
  else if (!TAf && TBf && OBF)
    k_gemm<false, true, true><<<g, blk, 0, st>>>(A, lda, sA, Bm, ldb, sB, bias, C, ldc, sC, M, N, K, alpha);
  else if (TAf && !TBf && OBF)
    k_gemm<true, false, true><<<g, blk, 0, st>>>(A, lda, sA, Bm, ldb, sB, bias, C, ldc, sC, M, N, K, alpha);
  else
    k_gemm<false, false, false><<<g, blk, 0, st>>>(A, lda, sA, Bm, ldb, sB, bias, C, ldc, sC, M, N, K, alpha);
}

static inline void cast_f(hipStream_t st, const float* s, ushort_t* d, long n) {
  k_cast<<<(unsigned)((n + 255) / 256), 256, 0, st>>>(s, d, n);
}

extern "C" void kernel_launch(void* const* d_in, const int* in_sizes, int n_in,
                              void* d_out, int out_size, void* d_ws, size_t ws_size,
                              hipStream_t stream) {
  const float* tokens   = (const float*)d_in[0];
  const float* cls_tok  = (const float*)d_in[1];
  const float* fnn1_w = (const float*)d_in[22];
  const float* fnn1_b = (const float*)d_in[23];
  const float* fnn2_w = (const float*)d_in[24];
  const float* fnn2_b = (const float*)d_in[25];

  // ---- carve workspace (bf16 = ushort elements), 256B aligned ----
  size_t off = 0;
  auto carve = [&](size_t elems) -> ushort_t* {
    ushort_t* p = (ushort_t*)d_ws + off;
    off += (elems + 127) & ~(size_t)127;
    return p;
  };
  const size_t XMAX = (size_t)B_ * LA_ * H_;        // max activation size
  ushort_t* X    = carve(XMAX);                      // (B, L, 768)
  ushort_t* XT   = carve(XMAX);                      // (B, 768, L)
  ushort_t* Q    = carve(XMAX);
  ushort_t* Kb   = carve(XMAX);
  ushort_t* V    = carve(XMAX);
  ushort_t* ATTN = carve((size_t)B_ * H_ * H_);      // scores / probs (bf16)
  ushort_t* SAT  = carve(XMAX);                      // token-attn output
  ushort_t* SAF  = carve(XMAX);                      // feature-attn output
  ushort_t* T100 = carve((size_t)B_ * LA_ * 100);
  ushort_t* F100 = carve((size_t)B_ * LA_ * 100);
  ushort_t* TF   = carve((size_t)B_ * 10000);        // (B,100,100)
  ushort_t* FTW  = carve((size_t)10000 * 1000);      // aft_w / bft_w bf16
  ushort_t* QKVW = carve((size_t)3 * H_ * H_);
  ushort_t* FQW  = carve((size_t)3 * LA_ * LA_);
  ushort_t* TWW  = carve((size_t)H_ * 100);
  ushort_t* FWW  = carve((size_t)H_ * 100);
  ushort_t* F1W  = carve((size_t)2 * H_ * H_);
  ushort_t* F2W  = carve((size_t)H_ * 4);
  ushort_t* HBF  = carve((size_t)B_ * 2 * H_);
  ushort_t* HEAD = carve((size_t)B_ * H_);
  (void)ws_size; (void)in_sizes; (void)n_in; (void)out_size;

  float* out = (float*)d_out;

  // ---- CLS head: (64x1536)@(1536x768) -> (64x768)@(768x3) -> out[:,0:3] ----
  k_build_h<<<(B_ * 2 * H_ + 255) / 256, 256, 0, stream>>>(tokens, cls_tok, HBF);
  cast_f(stream, fnn1_w, F1W, (long)2 * H_ * H_);
  cast_f(stream, fnn2_w, F2W, (long)H_ * 3);
  gemm(stream, false, false, true, HBF, 2 * H_, 0, F1W, H_, 0, fnn1_b,
       HEAD, H_, 0, B_, H_, 2 * H_, 1, 1.f);
  gemm(stream, false, false, false, HEAD, H_, 0, F2W, 3, 0, fnn2_b,
       out + 0, OUTC, 0, B_, 3, H_, 1, 1.f);

  // ---- the two attention branches (sequential; buffers reused) ----
  struct Br {
    const float* xsrc; int L;
    const float *qkvw, *qkvb, *fqkvw, *fqkvb, *tWw, *tWb, *fWw, *fWb, *ftw, *ftb;
    int outoff;
  };
  const Br br[2] = {
      {tokens, LA_, (const float*)d_in[2], (const float*)d_in[3],
       (const float*)d_in[4], (const float*)d_in[5], (const float*)d_in[10],
       (const float*)d_in[11], (const float*)d_in[12], (const float*)d_in[13],
       (const float*)d_in[18], (const float*)d_in[19], 3},
      {cls_tok, ML_, (const float*)d_in[6], (const float*)d_in[7],
       (const float*)d_in[8], (const float*)d_in[9], (const float*)d_in[14],
       (const float*)d_in[15], (const float*)d_in[16], (const float*)d_in[17],
       (const float*)d_in[20], (const float*)d_in[21], 1003},
  };

  for (int bi = 0; bi < 2; ++bi) {
    const Br& p = br[bi];
    const int  L  = p.L;
    const long LH = (long)L * H_;
    const float tscale = 1.f / sqrtf((float)H_);
    const float fscale = 1.f / sqrtf((float)L);

    // cast activations (+ transpose) and this branch's weights to bf16
    k_cast_xt<<<(unsigned)(((long)B_ * L * H_ + 255) / 256), 256, 0, stream>>>(
        p.xsrc, X, XT, L, (long)B_ * L * H_);
    cast_f(stream, p.qkvw, QKVW, (long)3 * H_ * H_);
    cast_f(stream, p.fqkvw, FQW, (long)3 * L * L);
    cast_f(stream, p.tWw, TWW, (long)H_ * 100);
    cast_f(stream, p.fWw, FWW, (long)H_ * 100);
    cast_f(stream, p.ftw, FTW, (long)10000 * 1000);

    // ---- token self-attention: (L x 768) ----
    for (int i = 0; i < 3; ++i) {
      ushort_t* dst = (i == 0) ? Q : (i == 1) ? Kb : V;
      gemm(stream, false, false, true, X, H_, LH, QKVW + (long)i * H_ * H_, H_, 0,
           p.qkvb + (long)i * H_, dst, H_, LH, L, H_, H_, B_, 1.f);
    }
    gemm(stream, false, true, true, Q, H_, LH, Kb, H_, LH, nullptr,
         ATTN, L, (long)L * L, L, L, H_, B_, tscale);
    k_softmax<<<B_ * L, 256, 0, stream>>>(ATTN, L);
    gemm(stream, false, false, true, ATTN, L, (long)L * L, V, H_, LH, nullptr,
         SAT, H_, LH, L, H_, L, B_, 1.f);

    // ---- feature self-attention: (768 x L) ----
    for (int i = 0; i < 3; ++i) {
      ushort_t* dst = (i == 0) ? Q : (i == 1) ? Kb : V;
      gemm(stream, false, false, true, XT, L, LH, FQW + (long)i * L * L, L, 0,
           p.fqkvb + (long)i * L, dst, L, LH, H_, L, L, B_, 1.f);
    }
    gemm(stream, false, true, true, Q, L, LH, Kb, L, LH, nullptr,
         ATTN, H_, (long)H_ * H_, H_, H_, L, B_, fscale);
    k_softmax<<<B_ * H_, 256, 0, stream>>>(ATTN, H_);
    gemm(stream, false, false, true, ATTN, H_, (long)H_ * H_, V, L, LH, nullptr,
         SAF, L, LH, H_, L, H_, B_, 1.f);

    // ---- 100-dim projections ----
    // aT = SAT @ tW + b          (L x 100)
    gemm(stream, false, false, true, SAT, H_, LH, TWW, 100, 0, p.tWb,
         T100, 100, (long)L * 100, L, 100, H_, B_, 1.f);
    // aF = SAF^T @ fW + b        (L x 100), SAF is (768 x L)
    gemm(stream, true, false, true, SAF, L, LH, FWW, 100, 0, p.fWb,
         F100, 100, (long)L * 100, L, 100, H_, B_, 1.f);

    // ---- bilinear: TF[m][n] = sum_l aT[l][m] * aF[l][n]  (100 x 100) ----
    gemm(stream, true, false, true, T100, 100, (long)L * 100, F100, 100,
         (long)L * 100, nullptr, TF, 100, 10000, 100, 100, L, B_, 1.f);

    // ---- readout: (64 x 10000) @ (10000 x 1000) + b -> out columns ----
    gemm(stream, false, false, false, TF, 10000, 0, FTW, 1000, 0, p.ftb,
         out + p.outoff, OUTC, 0, B_, 1000, 10000, 1, 1.f);
  }
}
